// net_17394617549299
// MI455X (gfx1250) — compile-verified, matrix operand order checked by reading, not run on hardware
//
#include <hip/hip_runtime.h>
#include <hip/hip_bf16.h>

#define NN 10000      // nodes
#define NE 160000     // edges
#define FIN 16
#define D 32
#define NG 64         // graphs
#define EH 128
#define DD 1024       // D*D
#define CDIV(a,b) (((a)+(b)-1)/(b))

typedef __attribute__((ext_vector_type(16))) _Float16 v16h;
typedef __attribute__((ext_vector_type(8)))  _Float16 v8h;
typedef __attribute__((ext_vector_type(8)))  float    v8f;

// ---- order-preserving float <-> uint encoding for atomic segment-max ----
__device__ __forceinline__ unsigned fenc(float f) {
  unsigned u = __float_as_uint(f);
  return (u & 0x80000000u) ? ~u : (u | 0x80000000u);
}
__device__ __forceinline__ float fdec(unsigned e) {
  return (e & 0x80000000u) ? __uint_as_float(e & 0x7fffffffu)
                           : __uint_as_float(~e);
}

// ---------------------------------------------------------------- fill ----
__global__ void k_fill_u32(unsigned* __restrict__ p, unsigned v, int n) {
  int i = blockIdx.x * blockDim.x + threadIdx.x;
  if (i < n) p[i] = v;
}

// ---------------- swizzle emlp_w2 (128x1024 f32) into WMMA B fragments ----
// Layout: frag f = nb*4 + kb ; per frag: [lane(32)][16 halves]
// half h of lane l holds B[k][n] with n = nb*16 + (l&15),
// k = kb*32 + (l<16?0:8) + (h<8 ? h : h+8)   (mirrors 16-bit A K-pattern)
__global__ void k_prep_w2(const float* __restrict__ W2, _Float16* __restrict__ Bf) {
  int id = blockIdx.x * blockDim.x + threadIdx.x;          // 0 .. 131071
  if (id >= EH * DD) return;
  int h    = id & 15;
  int lane = (id >> 4) & 31;
  int f    = id >> 9;
  int kb   = f & 3;
  int nb   = f >> 2;
  int k = kb * 32 + ((lane < 16) ? 0 : 8) + (h < 8 ? h : h + 8);
  int n = nb * 16 + (lane & 15);
  Bf[id] = (_Float16)W2[k * DD + n];
}

// ---------------- edge MLP layer 1: A = relu(edge_attr @ W1 + b1), f16 ----
__global__ void k_edge_mlp1(const float* __restrict__ ea, const float* __restrict__ W1,
                            const float* __restrict__ b1, _Float16* __restrict__ A) {
  int id = blockIdx.x * blockDim.x + threadIdx.x;          // NE*EH
  if (id >= NE * EH) return;
  int j = id & (EH - 1);
  int e = id >> 7;
  float v = b1[j] + ea[e * 3 + 0] * W1[0 * EH + j]
                  + ea[e * 3 + 1] * W1[1 * EH + j]
                  + ea[e * 3 + 2] * W1[2 * EH + j];
  A[id] = (_Float16)fmaxf(v, 0.0f);
}

// ---------------- WMMA GEMM: We = A(E x 128) @ W2(128 x 1024) + b2 -------
// Workgroup (4 waves) computes one 16-row x 256-col tile.
// A tile (16x128 f16 = 4KB) is staged once in LDS in fragment layout and
// shared by all 4 waves; each wave covers a different 64-col strip.
// grid = (E/16, 1024/256)
__global__ __launch_bounds__(128) void k_we_gemm(
    const _Float16* __restrict__ A,   // E x 128 row-major f16
    const _Float16* __restrict__ Bf,  // swizzled fragments
    const float*    __restrict__ b2,  // 1024
    _Float16*       __restrict__ We)  // E x 1024 f16
{
  __shared__ _Float16 sA[4 * 32 * 16];        // [kb][lane][16 halves] = 4KB

  const int wave = threadIdx.x >> 5;
  const int lane = threadIdx.x & 31;
  const int m0   = blockIdx.x * 16;

  // ---- cooperative stage of the A tile into LDS (fragment layout) ----
#pragma unroll
  for (int i = 0; i < 2; ++i) {
    int c     = threadIdx.x + i * 128;        // 256 chunks of 8 halves
    int hc    = c & 1;
    int sl    = (c >> 1) & 31;                // fragment lane
    int kb    = c >> 6;
    int base  = (sl < 16) ? 0 : 8;
    const _Float16* src = A + (size_t)(m0 + (sl & 15)) * EH + kb * 32 + base + hc * 16;
    *(v8h*)(sA + (kb * 32 + sl) * 16 + hc * 8) = *(const v8h*)src;
    // keep the A stream ahead: prefetch same chunk of the next m-tile
    __builtin_prefetch(src + 16 * EH, 0, 1);  // -> global_prefetch_b8
  }
  __syncthreads();

  const int n0 = blockIdx.y * 256 + wave * 64;

  v8f acc0 = {}, acc1 = {}, acc2 = {}, acc3 = {};
#pragma unroll
  for (int kb = 0; kb < 4; ++kb) {
    v16h a = *(const v16h*)(sA + (kb * 32 + lane) * 16);   // ds_load from LDS
    const int nbb = (n0 >> 4);
    v16h bfr0 = *(const v16h*)(Bf + (((nbb + 0) * 4 + kb) * 32 + lane) * 16);
    v16h bfr1 = *(const v16h*)(Bf + (((nbb + 1) * 4 + kb) * 32 + lane) * 16);
    v16h bfr2 = *(const v16h*)(Bf + (((nbb + 2) * 4 + kb) * 32 + lane) * 16);
    v16h bfr3 = *(const v16h*)(Bf + (((nbb + 3) * 4 + kb) * 32 + lane) * 16);
    acc0 = __builtin_amdgcn_wmma_f32_16x16x32_f16(false, a, false, bfr0, (short)0, acc0, false, false);
    acc1 = __builtin_amdgcn_wmma_f32_16x16x32_f16(false, a, false, bfr1, (short)0, acc1, false, false);
    acc2 = __builtin_amdgcn_wmma_f32_16x16x32_f16(false, a, false, bfr2, (short)0, acc2, false, false);
    acc3 = __builtin_amdgcn_wmma_f32_16x16x32_f16(false, a, false, bfr3, (short)0, acc3, false, false);
  }
  const int mrow = m0 + ((lane < 16) ? 0 : 8);
  const int ncol = lane & 15;
#pragma unroll
  for (int t = 0; t < 4; ++t) {
    const int n = n0 + t * 16 + ncol;
    const float bias = b2[n];
    const v8f acc = (t == 0) ? acc0 : (t == 1) ? acc1 : (t == 2) ? acc2 : acc3;
#pragma unroll
    for (int r = 0; r < 8; ++r)
      We[(size_t)(mrow + r) * DD + n] = (_Float16)(acc[r] + bias);
  }
}

// ---------------- h0 = relu(x @ lin0_w + lin0_b) --------------------------
__global__ void k_lin0(const float* __restrict__ x, const float* __restrict__ w,
                       const float* __restrict__ b, float* __restrict__ h) {
  int id = blockIdx.x * blockDim.x + threadIdx.x;
  if (id >= NN * D) return;
  int o = id & (D - 1), n = id >> 5;
  float v = b[o];
#pragma unroll
  for (int k = 0; k < FIN; ++k) v += x[n * FIN + k] * w[k * D + o];
  h[id] = fmaxf(v, 0.0f);
}

// ---------------- degree (segment count on dst) ---------------------------
__global__ void k_deg(const int* __restrict__ dst, float* __restrict__ deg) {
  int e = blockIdx.x * blockDim.x + threadIdx.x;
  if (e < NE) atomicAdd(deg + dst[e], 1.0f);
}

// ---------------- NNConv: per-edge message + scatter-add ------------------
// one wave per edge; lane = output channel
__global__ __launch_bounds__(256) void k_nnconv_edge(
    const _Float16* __restrict__ We, const float* __restrict__ h,
    const int* __restrict__ src, const int* __restrict__ dst,
    float* __restrict__ agg) {
  int wid = (blockIdx.x * blockDim.x + threadIdx.x) >> 5;
  int o   = threadIdx.x & 31;
  if (wid >= NE) return;
  int s = src[wid], d = dst[wid];
  const float* hs = h + s * D;
  const _Float16* w = We + (size_t)wid * DD + o;
  float m = 0.0f;
#pragma unroll
  for (int i = 0; i < D; ++i) m += hs[i] * (float)w[i * D];
  atomicAdd(agg + d * D + o, m);
}

// ---------------- NNConv node combine: mean + root + bias + relu ----------
__global__ void k_nnconv_node(const float* __restrict__ agg, const float* __restrict__ deg,
                              const float* __restrict__ h, const float* __restrict__ rw,
                              const float* __restrict__ cb, float* __restrict__ out) {
  int id = blockIdx.x * blockDim.x + threadIdx.x;
  if (id >= NN * D) return;
  int o = id & (D - 1), n = id >> 5;
  float v = agg[id] / fmaxf(deg[n], 1.0f) + cb[o];
#pragma unroll
  for (int i = 0; i < D; ++i) v += h[n * D + i] * rw[i * D + o];
  out[id] = fmaxf(v, 0.0f);
}

// ---------------- GAT: hw = h @ gat_w ; also zero scatter buffer ----------
__global__ void k_gat_hw(const float* __restrict__ h, const float* __restrict__ gw,
                         float* __restrict__ hw, float* __restrict__ acc) {
  int id = blockIdx.x * blockDim.x + threadIdx.x;
  if (id >= NN * D) return;
  int o = id & (D - 1), n = id >> 5;
  float v = 0.0f;
#pragma unroll
  for (int i = 0; i < D; ++i) v += h[n * D + i] * gw[i * D + o];
  hw[id]  = v;
  acc[id] = 0.0f;
}

// ---------------- GAT per-node: attention dots + segment init -------------
__global__ void k_gat_node(const float* __restrict__ hw, const float* __restrict__ as,
                           const float* __restrict__ ad, float* __restrict__ asum,
                           float* __restrict__ dsum, unsigned* __restrict__ mx,
                           float* __restrict__ den) {
  int n = blockIdx.x * blockDim.x + threadIdx.x;
  if (n >= NN) return;
  float a = 0.0f, b = 0.0f;
#pragma unroll
  for (int o = 0; o < D; ++o) { a += hw[n * D + o] * as[o]; b += hw[n * D + o] * ad[o]; }
  asum[n] = a; dsum[n] = b;
  mx[n] = fenc(-3.0e38f);
  den[n] = 0.0f;
}

__device__ __forceinline__ void gat_edge_idx(int j, const int* src, const int* dst,
                                             int& s, int& d) {
  if (j < NE) { s = src[j]; d = dst[j]; } else { s = j - NE; d = j - NE; }
}

// ---------------- GAT segment-max of leaky logits -------------------------
__global__ void k_gat_max(const int* __restrict__ src, const int* __restrict__ dst,
                          const float* __restrict__ asum, const float* __restrict__ dsum,
                          unsigned* __restrict__ mx) {
  int j = blockIdx.x * blockDim.x + threadIdx.x;
  if (j >= NE + NN) return;
  int s, d; gat_edge_idx(j, src, dst, s, d);
  float l = asum[s] + dsum[d];
  l = (l > 0.0f) ? l : 0.2f * l;
  atomicMax(mx + d, fenc(l));
}

// ---------------- GAT exp + segment-sum -----------------------------------
__global__ void k_gat_exp(const int* __restrict__ src, const int* __restrict__ dst,
                          const float* __restrict__ asum, const float* __restrict__ dsum,
                          const unsigned* __restrict__ mx, float* __restrict__ wexp,
                          float* __restrict__ den) {
  int j = blockIdx.x * blockDim.x + threadIdx.x;
  if (j >= NE + NN) return;
  int s, d; gat_edge_idx(j, src, dst, s, d);
  float l = asum[s] + dsum[d];
  l = (l > 0.0f) ? l : 0.2f * l;
  float w = __expf(l - fdec(mx[d]));
  wexp[j] = w;
  atomicAdd(den + d, w);
}

// ---------------- GAT weighted scatter (wave per edge) --------------------
__global__ __launch_bounds__(256) void k_gat_scatter(
    const int* __restrict__ src, const int* __restrict__ dst,
    const float* __restrict__ wexp, const float* __restrict__ den,
    const float* __restrict__ hw, float* __restrict__ acc) {
  int j = (blockIdx.x * blockDim.x + threadIdx.x) >> 5;
  int o = threadIdx.x & 31;
  if (j >= NE + NN) return;
  int s, d; gat_edge_idx(j, src, dst, s, d);
  float a = wexp[j] / den[d];
  atomicAdd(acc + d * D + o, a * hw[s * D + o]);
}

// ---------------- GAT output: bias + relu ---------------------------------
__global__ void k_gat_out(const float* __restrict__ acc, const float* __restrict__ gb,
                          float* __restrict__ m) {
  int id = blockIdx.x * blockDim.x + threadIdx.x;
  if (id >= NN * D) return;
  m[id] = fmaxf(acc[id] + gb[id & (D - 1)], 0.0f);
}

// ---------------- mean pool by graph --------------------------------------
__global__ void k_pool_cnt(const int* __restrict__ bid, float* __restrict__ gcnt) {
  int n = blockIdx.x * blockDim.x + threadIdx.x;
  if (n < NN) atomicAdd(gcnt + bid[n], 1.0f);
}
__global__ void k_pool_sum(const float* __restrict__ m1, const float* __restrict__ m3,
                           const float* __restrict__ m5, const int* __restrict__ bid,
                           float* __restrict__ pooled) {
  int id = blockIdx.x * blockDim.x + threadIdx.x;
  if (id >= NN * D) return;
  int o = id & (D - 1), n = id >> 5;
  float v = (m1[id] + m3[id] + m5[id]) * (1.0f / 3.0f);
  atomicAdd(pooled + bid[n] * D + o, v);
}
__global__ void k_lin1(const float* __restrict__ pooled, const float* __restrict__ gcnt,
                       const float* __restrict__ w, const float* __restrict__ b,
                       float* __restrict__ p2) {
  int id = blockIdx.x * blockDim.x + threadIdx.x;
  if (id >= NG * D) return;
  int o = id & (D - 1), g = id >> 5;
  float inv = 1.0f / fmaxf(gcnt[g], 1.0f);
  float v = b[o];
#pragma unroll
  for (int i = 0; i < D; ++i) v += pooled[g * D + i] * inv * w[i * D + o];
  p2[id] = fmaxf(v, 0.0f);
}
__global__ void k_lin2(const float* __restrict__ p2, const float* __restrict__ w,
                       const float* __restrict__ b, float* __restrict__ out) {
  int g = blockIdx.x * blockDim.x + threadIdx.x;
  if (g >= NG) return;
  float v = b[0];
#pragma unroll
  for (int o = 0; o < D; ++o) v += p2[g * D + o] * w[o];
  out[g] = v;
}

// ==========================================================================
extern "C" void kernel_launch(void* const* d_in, const int* in_sizes, int n_in,
                              void* d_out, int out_size, void* d_ws, size_t ws_size,
                              hipStream_t stream) {
  const float* x       = (const float*)d_in[0];
  const int*   ei      = (const int*)  d_in[1];
  const float* eattr   = (const float*)d_in[2];
  const int*   bid     = (const int*)  d_in[3];
  const float* lin0_w  = (const float*)d_in[4];
  const float* lin0_b  = (const float*)d_in[5];
  const float* emlp_w1 = (const float*)d_in[6];
  const float* emlp_b1 = (const float*)d_in[7];
  const float* emlp_w2 = (const float*)d_in[8];
  const float* emlp_b2 = (const float*)d_in[9];
  const float* root_w  = (const float*)d_in[10];
  const float* conv_b  = (const float*)d_in[11];
  const float* gat_w   = (const float*)d_in[12];
  const float* gat_as  = (const float*)d_in[13];
  const float* gat_ad  = (const float*)d_in[14];
  const float* gat_b   = (const float*)d_in[15];
  const float* lin1_w  = (const float*)d_in[16];
  const float* lin1_b  = (const float*)d_in[17];
  const float* lin2_w  = (const float*)d_in[18];
  const float* lin2_b  = (const float*)d_in[19];
  const int* src = ei;
  const int* dst = ei + NE;
  float* out = (float*)d_out;

  // ---- workspace carve-up (256B aligned) ----
  char* ws = (char*)d_ws;
  size_t off = 0;
  auto alloc = [&](size_t bytes) -> void* {
    void* p = ws + off;
    off += (bytes + 255) & ~(size_t)255;
    return p;
  };
  _Float16* We   = (_Float16*)alloc((size_t)NE * DD * 2);   // 327.7 MB
  _Float16* Ah   = (_Float16*)alloc((size_t)NE * EH * 2);   // 41 MB
  _Float16* Bf   = (_Float16*)alloc((size_t)EH * DD * 2);
  float* h0      = (float*)alloc((size_t)NN * D * 4);
  float* tnn     = (float*)alloc((size_t)NN * D * 4);
  float* mbuf0   = (float*)alloc((size_t)NN * D * 4);
  float* mbuf1   = (float*)alloc((size_t)NN * D * 4);
  float* mbuf2   = (float*)alloc((size_t)NN * D * 4);
  float* hw      = (float*)alloc((size_t)NN * D * 4);
  float* agg     = (float*)alloc((size_t)NN * D * 4);
  float* gacc    = (float*)alloc((size_t)NN * D * 4);
  float* deg     = (float*)alloc((size_t)NN * 4);
  float* asum    = (float*)alloc((size_t)NN * 4);
  float* dsum    = (float*)alloc((size_t)NN * 4);
  unsigned* mx   = (unsigned*)alloc((size_t)NN * 4);
  float* den     = (float*)alloc((size_t)NN * 4);
  float* wexp    = (float*)alloc((size_t)(NE + NN) * 4);
  float* pooled  = (float*)alloc((size_t)NG * D * 4);
  float* p2      = (float*)alloc((size_t)NG * D * 4);
  float* gcnt    = (float*)alloc((size_t)NG * 4);
  float* mbufs[3] = {mbuf0, mbuf1, mbuf2};

  // ---- edge MLP (shared across layers): A f16, then WMMA GEMM -> We f16 --
  k_prep_w2   <<<CDIV(EH * DD, 256), 256, 0, stream>>>(emlp_w2, Bf);
  k_edge_mlp1 <<<CDIV(NE * EH, 256), 256, 0, stream>>>(eattr, emlp_w1, emlp_b1, Ah);
  k_we_gemm   <<<dim3(NE / 16, DD / 256), 128, 0, stream>>>(Ah, Bf, emlp_b2, We);

  // ---- h0 and degree ----
  k_lin0      <<<CDIV(NN * D, 256), 256, 0, stream>>>(x, lin0_w, lin0_b, h0);
  k_fill_u32  <<<CDIV(NN, 256), 256, 0, stream>>>((unsigned*)deg, 0u, NN);
  k_deg       <<<CDIV(NE, 256), 256, 0, stream>>>(dst, deg);

  // ---- 3 x (NNConv -> ReLU -> GAT -> ReLU) ----
  for (int l = 0; l < 3; ++l) {
    const float* hin = (l == 0) ? h0 : mbufs[l - 1];
    k_fill_u32    <<<CDIV(NN * D, 256), 256, 0, stream>>>((unsigned*)agg, 0u, NN * D);
    k_nnconv_edge <<<CDIV(NE * 32, 256), 256, 0, stream>>>(We, hin, src, dst, agg);
    k_nnconv_node <<<CDIV(NN * D, 256), 256, 0, stream>>>(agg, deg, hin, root_w, conv_b, tnn);
    k_gat_hw      <<<CDIV(NN * D, 256), 256, 0, stream>>>(tnn, gat_w, hw, gacc);
    k_gat_node    <<<CDIV(NN, 256), 256, 0, stream>>>(hw, gat_as, gat_ad, asum, dsum, mx, den);
    k_gat_max     <<<CDIV(NE + NN, 256), 256, 0, stream>>>(src, dst, asum, dsum, mx);
    k_gat_exp     <<<CDIV(NE + NN, 256), 256, 0, stream>>>(src, dst, asum, dsum, mx, wexp, den);
    k_gat_scatter <<<CDIV((NE + NN) * 32, 256), 256, 0, stream>>>(src, dst, wexp, den, hw, gacc);
    k_gat_out     <<<CDIV(NN * D, 256), 256, 0, stream>>>(gacc, gat_b, mbufs[l]);
  }

  // ---- mean pool per graph + head MLP ----
  k_fill_u32 <<<CDIV(NG * D, 256), 256, 0, stream>>>((unsigned*)pooled, 0u, NG * D);
  k_fill_u32 <<<1, 256, 0, stream>>>((unsigned*)gcnt, 0u, NG);
  k_pool_cnt <<<CDIV(NN, 256), 256, 0, stream>>>(bid, gcnt);
  k_pool_sum <<<CDIV(NN * D, 256), 256, 0, stream>>>(mbuf0, mbuf1, mbuf2, bid, pooled);
  k_lin1     <<<CDIV(NG * D, 256), 256, 0, stream>>>(pooled, gcnt, lin1_w, lin1_b, p2);
  k_lin2     <<<1, NG, 0, stream>>>(p2, lin2_w, lin2_b, out);
}
